// RobustGeoSNN_59828894433363
// MI455X (gfx1250) — compile-verified, matrix operand order checked by reading, not run on hardware
//
#include <hip/hip_runtime.h>
#include <hip/hip_bf16.h>

// ---------------------------------------------------------------------------
// RobustGeoSNN forward on gfx1250 (MI455X).
// Main cost: stream 131MB of kin_spikes once; fuse the [B*T,64]x[64,128] GEMM
// into the T=1000 LIF scan using v_wmma_f32_16x16x32_bf16 (f32 accumulate).
// Global loads are software-pipelined 2 deep so each iteration consumes data
// loaded a full iteration earlier; cache prefetch runs 4 steps ahead.
// ---------------------------------------------------------------------------

#define Bsz   512
#define Tlen  1000
#define Cin   64
#define Hd    128
#define TDAd  150
#define NCLS  4

typedef __bf16 v16bf __attribute__((ext_vector_type(16)));
typedef __bf16 v8bf  __attribute__((ext_vector_type(8)));
typedef __bf16 v4bf  __attribute__((ext_vector_type(4)));
typedef float  v8f   __attribute__((ext_vector_type(8)));

union Frag {
    v16bf v;
    v8bf  h[2];
    __bf16 e[16];
};

// ---------------------------------------------------------------------------
// Kernel 1: fused currents-GEMM + LIF scan.
// grid = 32 blocks (16 batches each), block = 256 threads = 8 waves (wave32).
// Wave w owns H columns [16w, 16w+16). mem/cnt live in WMMA C/D layout.
// ---------------------------------------------------------------------------
__global__ __launch_bounds__(256)
void snn_scan_kernel(const float* __restrict__ x,      // [B,T,C]
                     const float* __restrict__ W_fc,   // [H,C]
                     const float* __restrict__ b_fc,   // [H]
                     float* __restrict__ counts_out)   // [B,H]
{
    __shared__ __align__(16) __bf16 xs[2][16][Cin];    // double-buffered x_t tile (bf16)

    const int tid  = threadIdx.x;
    const int wave = tid >> 5;
    const int lane = tid & 31;
    const int b0   = blockIdx.x * 16;

    const int row  = lane & 15;                 // A-row / D-column index
    const int kb   = (lane < 16) ? 0 : 8;       // 16-bit fragment K-base for this lane
    const int hcol = wave * 16 + row;           // this lane's B/D column (H index)

    // --- B fragments: W_fc^T slice [64 x 16] for this wave, bf16, loaded once ---
    Frag fb0, fb1;
    const float* wr = W_fc + (size_t)hcol * Cin;
#pragma unroll
    for (int j = 0; j < 8; ++j) {
        fb0.e[j]     = (__bf16)wr[kb + j];
        fb0.e[8 + j] = (__bf16)wr[kb + 16 + j];
        fb1.e[j]     = (__bf16)wr[kb + 32 + j];
        fb1.e[8 + j] = (__bf16)wr[kb + 48 + j];
    }
    const float bias = b_fc[hcol];

    v8f mem = {};
    v8f cnt = {};

    // --- cooperative x-tile loader: 1024 elems / 256 threads = one float4 each ---
    const int lrow = tid >> 4;                  // 0..15 (batch row in tile)
    const int lcol = (tid & 15) * 4;            // 0..60 (C offset)
    const float* srcbase =
        x + ((size_t)(b0 + lrow) * Tlen) * Cin + lcol;

    // preload t = 0 into LDS; start the t = 1 load into registers (2-deep pipe)
    {
        float4 v = *(const float4*)srcbase;
        v4bf pv = { (__bf16)v.x, (__bf16)v.y, (__bf16)v.z, (__bf16)v.w };
        *(v4bf*)&xs[0][lrow][lcol] = pv;
    }
    float4 vn = *(const float4*)(srcbase + (size_t)1 * Cin);   // x(t+1) in regs
    __syncthreads();

    for (int t = 0; t < Tlen; ++t) {
        const int p = t & 1;

        // issue global load for t+2 now; consume it one iteration from now.
        float4 vn2;
        if (t + 2 < Tlen) {
            vn2 = *(const float4*)(srcbase + (size_t)(t + 2) * Cin);
            if (t + 4 < Tlen)
                __builtin_prefetch(srcbase + (size_t)(t + 4) * Cin, 0, 1);
        }

        // A fragments from LDS (bf16, matches 16-bit A VGPR layout)
        Frag a0, a1;
        a0.h[0] = *(const v8bf*)&xs[p][row][kb];
        a0.h[1] = *(const v8bf*)&xs[p][row][kb + 16];
        a1.h[0] = *(const v8bf*)&xs[p][row][kb + 32];
        a1.h[1] = *(const v8bf*)&xs[p][row][kb + 48];

        // mem' = 0.9*mem + bias + x_t @ W_fc^T   (K = 64 -> two chained WMMAs)
        v8f c = mem * 0.9f + bias;
        c = __builtin_amdgcn_wmma_f32_16x16x32_bf16(false, a0.v, false, fb0.v,
                                                    (short)0, c, false, false);
        c = __builtin_amdgcn_wmma_f32_16x16x32_bf16(false, a1.v, false, fb1.v,
                                                    (short)0, c, false, false);

        // threshold / reset / count (elementwise, layout-agnostic)
#pragma unroll
        for (int i = 0; i < 8; ++i) {
            float m = c[i];
            bool fire = (m >= 1.0f);
            cnt[i] += fire ? 1.0f : 0.0f;
            mem[i] = fire ? 0.0f : m;
        }

        // store x(t+1) (loaded last iteration, already resident) into other buffer
        if (t + 1 < Tlen) {
            v4bf pv = { (__bf16)vn.x, (__bf16)vn.y, (__bf16)vn.z, (__bf16)vn.w };
            *(v4bf*)&xs[p ^ 1][lrow][lcol] = pv;
        }
        vn = vn2;   // rotate the pipeline (garbage past the end is never stored)
        __syncthreads();
    }

    // write spike counts: D layout -> lanes 0-15: M=r, lanes 16-31: M=8+r
#pragma unroll
    for (int r = 0; r < 8; ++r) {
        int m = r + ((lane >> 4) << 3);
        counts_out[(size_t)(b0 + m) * Hd + hcol] = cnt[r];
    }
}

// ---------------------------------------------------------------------------
// Kernel 2: TDA MLP. grid = 512 (one block per batch), block = 64.
// ---------------------------------------------------------------------------
__global__ __launch_bounds__(64)
void tda_kernel(const float* __restrict__ tda, const float* __restrict__ W_t1,
                const float* __restrict__ b_t1, const float* __restrict__ W_t2,
                const float* __restrict__ b_t2, float* __restrict__ tda_out)
{
    __shared__ float sin_[TDAd];
    __shared__ float sh[64];
    const int b = blockIdx.x, tid = threadIdx.x;

    for (int k = tid; k < TDAd; k += 64) sin_[k] = tda[(size_t)b * TDAd + k];
    __syncthreads();

    float acc = b_t1[tid];
    const float* w = W_t1 + (size_t)tid * TDAd;
    for (int k = 0; k < TDAd; ++k) acc = fmaf(sin_[k], w[k], acc);
    sh[tid] = fmaxf(acc, 0.0f);
    __syncthreads();

    float acc2 = b_t2[tid];
    const float* w2 = W_t2 + (size_t)tid * 64;
#pragma unroll 8
    for (int k = 0; k < 64; ++k) acc2 = fmaf(sh[k], w2[k], acc2);
    tda_out[(size_t)b * 64 + tid] = fmaxf(acc2, 0.0f);
}

// ---------------------------------------------------------------------------
// Kernel 3: h = [counts/T || tda_out] @ W_c1^T + b_c1.  grid = 512, block = 128.
// ---------------------------------------------------------------------------
__global__ __launch_bounds__(128)
void fc1_kernel(const float* __restrict__ counts, const float* __restrict__ tda_out,
                const float* __restrict__ W_c1, const float* __restrict__ b_c1,
                float* __restrict__ h)
{
    __shared__ float fused[Hd + 64];
    const int b = blockIdx.x, f = threadIdx.x;

    fused[f] = counts[(size_t)b * Hd + f] * (1.0f / (float)Tlen);
    if (f < 64) fused[Hd + f] = tda_out[(size_t)b * 64 + f];
    __syncthreads();

    float acc = b_c1[f];
    const float* w = W_c1 + (size_t)f * (Hd + 64);
#pragma unroll 8
    for (int k = 0; k < Hd + 64; ++k) acc = fmaf(fused[k], w[k], acc);
    h[(size_t)b * Hd + f] = acc;
}

// ---------------------------------------------------------------------------
// Kernel 4: batch-norm statistics (training-mode, biased var) folded into
// scale/shift. grid = 128 (one block per feature), block = 256.
// ---------------------------------------------------------------------------
__global__ __launch_bounds__(256)
void bnstat_kernel(const float* __restrict__ h, const float* __restrict__ gamma,
                   const float* __restrict__ beta, float* __restrict__ scale,
                   float* __restrict__ shift)
{
    __shared__ float rs[256];
    __shared__ float rq[256];
    const int f = blockIdx.x, tid = threadIdx.x;

    float a = h[(size_t)tid * Hd + f];
    float c = h[(size_t)(tid + 256) * Hd + f];
    rs[tid] = a + c;
    rq[tid] = a * a + c * c;
    __syncthreads();

    for (int s = 128; s > 0; s >>= 1) {
        if (tid < s) { rs[tid] += rs[tid + s]; rq[tid] += rq[tid + s]; }
        __syncthreads();
    }
    if (tid == 0) {
        float mean = rs[0] * (1.0f / (float)Bsz);
        float var  = rq[0] * (1.0f / (float)Bsz) - mean * mean;
        float sc   = gamma[f] * rsqrtf(var + 1e-5f);
        scale[f] = sc;
        shift[f] = beta[f] - mean * sc;
    }
}

// ---------------------------------------------------------------------------
// Kernel 5: normalize + relu + final [128 -> 4] GEMV.  grid = 512, block = 128.
// ---------------------------------------------------------------------------
__global__ __launch_bounds__(128)
void fc2_kernel(const float* __restrict__ h, const float* __restrict__ scale,
                const float* __restrict__ shift, const float* __restrict__ W_c2,
                const float* __restrict__ b_c2, float* __restrict__ out)
{
    __shared__ float hn[Hd];
    const int b = blockIdx.x, f = threadIdx.x;

    hn[f] = fmaxf(fmaf(h[(size_t)b * Hd + f], scale[f], shift[f]), 0.0f);
    __syncthreads();

    if (f < NCLS) {
        float acc = b_c2[f];
        const float* w = W_c2 + (size_t)f * Hd;
#pragma unroll 8
        for (int k = 0; k < Hd; ++k) acc = fmaf(hn[k], w[k], acc);
        out[(size_t)b * NCLS + f] = acc;
    }
}

// ---------------------------------------------------------------------------
extern "C" void kernel_launch(void* const* d_in, const int* in_sizes, int n_in,
                              void* d_out, int out_size, void* d_ws, size_t ws_size,
                              hipStream_t stream)
{
    (void)in_sizes; (void)n_in; (void)out_size; (void)ws_size;

    const float* x     = (const float*)d_in[0];   // kin_spikes_seq [B,T,C]
    const float* tda   = (const float*)d_in[1];   // [B,TDA]
    const float* W_fc  = (const float*)d_in[2];   // [H,C]
    const float* b_fc  = (const float*)d_in[3];   // [H]
    const float* W_t1  = (const float*)d_in[4];   // [64,TDA]
    const float* b_t1  = (const float*)d_in[5];
    const float* W_t2  = (const float*)d_in[6];   // [64,64]
    const float* b_t2  = (const float*)d_in[7];
    const float* W_c1  = (const float*)d_in[8];   // [128,192]
    const float* b_c1  = (const float*)d_in[9];
    const float* gamma = (const float*)d_in[10];  // [128]
    const float* beta  = (const float*)d_in[11];  // [128]
    const float* W_c2  = (const float*)d_in[12];  // [4,128]
    const float* b_c2  = (const float*)d_in[13];

    float* out    = (float*)d_out;                // [B,NCLS]
    float* counts = out + (size_t)Bsz * NCLS;     // [B,H]  (second tuple output)

    float* ws      = (float*)d_ws;
    float* tda_out = ws;                          // B*64
    float* hbuf    = tda_out + (size_t)Bsz * 64;  // B*128
    float* scale   = hbuf + (size_t)Bsz * Hd;     // 128
    float* shift   = scale + Hd;                  // 128

    snn_scan_kernel<<<Bsz / 16, 256, 0, stream>>>(x, W_fc, b_fc, counts);
    tda_kernel<<<Bsz, 64, 0, stream>>>(tda, W_t1, b_t1, W_t2, b_t2, tda_out);
    fc1_kernel<<<Bsz, 128, 0, stream>>>(counts, tda_out, W_c1, b_c1, hbuf);
    bnstat_kernel<<<Hd, 256, 0, stream>>>(hbuf, gamma, beta, scale, shift);
    fc2_kernel<<<Bsz, 128, 0, stream>>>(hbuf, scale, shift, W_c2, b_c2, out);
}
